// TropicalLinear_85615878078631
// MI455X (gfx1250) — matrix operand order repaired and checked by reading, not compiled, not run on hardware
//
#include <hip/hip_runtime.h>

// Tropical linear: out[row,o] = m[row] + log( sum_k exp(x[row,k]-m[row]) * exp(w[k,o]) )
// row = b*N+n, ROWS = 8*2048 = 16384, K = IN = 128, O = OUT = 128.
// Inner sum is a GEMM in exp-space -> v_wmma_f32_16x16x32_f16.

typedef _Float16 v16h __attribute__((ext_vector_type(16)));
typedef _Float16 v8h  __attribute__((ext_vector_type(8)));
typedef float    v8f  __attribute__((ext_vector_type(8)));
typedef int      v4i  __attribute__((ext_vector_type(4)));

#define K_DIM   128
#define O_DIM   128
#define ROWS    (8 * 2048)

// ---------------------------------------------------------------------------
// Prep: EwT[n][k] = (f16) exp(w[k][n])   (transposed so B-fragments read
// contiguous K). 16384 elements, trivial.
// ---------------------------------------------------------------------------
__global__ void tl_prep_w(const float* __restrict__ w, _Float16* __restrict__ ewt) {
    int idx = blockIdx.x * blockDim.x + threadIdx.x;   // idx = k*128 + n
    if (idx < K_DIM * O_DIM) {
        int n = idx & (O_DIM - 1);
        int k = idx >> 7;
        ewt[n * K_DIM + k] = (_Float16)__expf(w[idx]); // coalesced read of w
    }
}

// ---------------------------------------------------------------------------
// Fused: per-row max -> exp -> WMMA GEMM vs LDS-staged EwT -> log epilogue.
// Block = 128 threads (4 waves); each wave computes a 16x128 output strip.
// ---------------------------------------------------------------------------
__global__ void __launch_bounds__(128) tl_fused_gemm(
    const float* __restrict__ x,          // [ROWS, K_DIM] f32
    const _Float16* __restrict__ ewt,     // [O_DIM, K_DIM] f16 (n-major)
    float* __restrict__ out)              // [ROWS, O_DIM] f32
{
    __shared__ _Float16 lds_ewt[O_DIM * K_DIM];       // 32 KB

    const int wave    = threadIdx.x >> 5;
    const int lane    = threadIdx.x & 31;
    const int half    = lane >> 4;                    // 0: lanes 0-15, 1: lanes 16-31
    const int r       = lane & 15;
    const int rowBase = (blockIdx.x * 4 + wave) * 16;

    // ---- Kick off LDS staging of exp(w)^T (32 KB per workgroup).
    // Prefer the CDNA5 async-to-LDS path (ASYNCcnt, no VGPR round trip);
    // fall back to a plain uint4 copy if the builtin isn't available.
    const int total_chunks = (O_DIM * K_DIM) / 8;     // 2048 x 16B
#if __has_builtin(__builtin_amdgcn_global_load_async_to_lds_b128)
    {
        typedef __attribute__((address_space(1))) v4i g_v4i;   // global
        typedef __attribute__((address_space(3))) v4i l_v4i;   // LDS
        const v4i* src = (const v4i*)ewt;
        v4i*       dst = (v4i*)lds_ewt;
        for (int i = threadIdx.x; i < total_chunks; i += blockDim.x) {
            __builtin_amdgcn_global_load_async_to_lds_b128(
                (g_v4i*)(src + i),                 // global source (AS1)
                (l_v4i*)(dst + i),                 // LDS destination (AS3)
                /*offset=*/0, /*cpol=*/0);
        }
    }
#else
    {
        const uint4* src = (const uint4*)ewt;
        uint4*       dst = (uint4*)lds_ewt;
        for (int i = threadIdx.x; i < total_chunks; i += blockDim.x)
            dst[i] = src[i];
    }
#endif

    // ---- Overlap with staging: load this lane's half-row of x in A-fragment
    // order and track the running max. (Independent of LDS.)
    // 16-bit A 16x32 layout: lane group `half`, element i<8 -> K = k0+8*half+i,
    // element i>=8 -> K = k0+16+8*half+(i-8).
    const float* xr = x + (size_t)(rowBase + r) * K_DIM;

    float xv[64];
    float m = -3.0e38f;
#pragma unroll
    for (int s = 0; s < 4; ++s) {
        const int k0 = s * 32;
#pragma unroll
        for (int i = 0; i < 8; ++i) {
            float a = xr[k0 + 8 * half + i];
            float b = xr[k0 + 16 + 8 * half + i];
            xv[s * 16 + i]     = a;
            xv[s * 16 + 8 + i] = b;
            m = fmaxf(m, fmaxf(a, b));
        }
    }
    // Lanes L and L^16 hold complementary K-halves of the same row -> combine.
    m = fmaxf(m, __shfl_xor(m, 16, 32));              // m = rowmax of row rowBase+r

    // ---- exp(x - m) -> f16 A fragments (values in (0,1], safe in f16).
    v16h afrag[4];
#pragma unroll
    for (int s = 0; s < 4; ++s) {
#pragma unroll
        for (int i = 0; i < 16; ++i)
            afrag[s][i] = (_Float16)__expf(xv[s * 16 + i] - m);
    }

    // ---- Complete staging: wait for this wave's async transfers, then make
    // all waves' staging visible.
#if __has_builtin(__builtin_amdgcn_global_load_async_to_lds_b128)
#if __has_builtin(__builtin_amdgcn_s_wait_asynccnt)
    __builtin_amdgcn_s_wait_asynccnt(0);
#else
    asm volatile("s_wait_asynccnt 0" ::: "memory");
#endif
#endif
    __syncthreads();

    // ---- WMMA: 4 K-steps x 8 N-tiles, f32 accumulators.
    v8f acc[8];
    const v8f zero = {0.f, 0.f, 0.f, 0.f, 0.f, 0.f, 0.f, 0.f};
#pragma unroll
    for (int t = 0; t < 8; ++t) acc[t] = zero;

#pragma unroll
    for (int s = 0; s < 4; ++s) {
        const int k0 = s * 32;
#pragma unroll
        for (int t = 0; t < 8; ++t) {
            // 16-bit B 32x16 layout: lane -> col (lane&15); element i<8 -> K = k0+8*half+i,
            // element i>=8 -> K = k0+16+8*half+(i-8) (matches A).
            const _Float16* bp = &lds_ewt[(t * 16 + r) * K_DIM + k0 + 8 * half];
            v8h blo = *(const v8h*)bp;
            v8h bhi = *(const v8h*)(bp + 16);
            v16h bfrag = __builtin_shufflevector(blo, bhi,
                0, 1, 2, 3, 4, 5, 6, 7, 8, 9, 10, 11, 12, 13, 14, 15);
            acc[t] = __builtin_amdgcn_wmma_f32_16x16x32_f16(
                /*neg_a=*/false, afrag[s], /*neg_b=*/false, bfrag,
                /*c_mod=*/(short)0, acc[t], /*reuse_a=*/false, /*reuse_b=*/false);
        }
    }

    // ---- Epilogue: out = m_row + log(acc).
    // C/D layout: lane -> col (lane&15); VGPR vr -> row vr + 8*half.
    // Row-max for output row rowBase+8*half+vr lives in lane (8*half+vr).
    float mv[8];
#pragma unroll
    for (int vr = 0; vr < 8; ++vr)
        mv[vr] = __shfl(m, 8 * half + vr, 32);

#pragma unroll
    for (int t = 0; t < 8; ++t) {
#pragma unroll
        for (int vr = 0; vr < 8; ++vr) {
            const int row = rowBase + 8 * half + vr;
            out[(size_t)row * O_DIM + t * 16 + r] = mv[vr] + __logf(acc[t][vr]);
        }
    }
}

// ---------------------------------------------------------------------------
extern "C" void kernel_launch(void* const* d_in, const int* in_sizes, int n_in,
                              void* d_out, int out_size, void* d_ws, size_t ws_size,
                              hipStream_t stream) {
    const float* x = (const float*)d_in[0];   // [8,2048,128] f32
    const float* w = (const float*)d_in[1];   // [128,128]    f32
    float* out     = (float*)d_out;           // [8,2048,128] f32

    _Float16* ewt = (_Float16*)d_ws;          // 32 KB scratch: exp(w)^T in f16

    tl_prep_w<<<(K_DIM * O_DIM + 255) / 256, 256, 0, stream>>>(w, ewt);
    tl_fused_gemm<<<ROWS / 64, 128, 0, stream>>>(x, ewt, out);
}
